// SelfAttention2d_16612933501606
// MI455X (gfx1250) — matrix operand order repaired and checked by reading, not compiled, not run on hardware
//
#include <hip/hip_runtime.h>

typedef __attribute__((ext_vector_type(16))) _Float16 v16h;
typedef __attribute__((ext_vector_type(8)))  _Float16 v8h;
typedef __attribute__((ext_vector_type(8)))  float    v8f;

static constexpr int Bn = 8;     // batch
static constexpr int C  = 512;   // channels
static constexpr int CK = 64;    // att channels
static constexpr int N  = 4096;  // H*W

#define WMMA_F16(a, b, c) \
  __builtin_amdgcn_wmma_f32_16x16x32_f16(false, (a), false, (b), (short)0, (c), false, false)

union HU { v16h v; v8h h[2]; };

// A operand 16x32 f16 from row-major [rows][ld]:
// lanes 0-15: M=lane, K={0..7,16..23}; lanes 16-31: M=lane-16, K={8..15,24..31}
__device__ __forceinline__ v16h load_a16(const _Float16* base, int ld, int lane) {
  int r  = lane & 15;
  int kh = (lane & 16) ? 8 : 0;
  HU u;
  u.h[0] = *(const v8h*)(base + r * ld + kh);
  u.h[1] = *(const v8h*)(base + r * ld + kh + 16);
  return u.v;
}

// B operand 32x16 f16 fed as B^T rows (lane = output col, K contiguous):
// lanes 0-15: K={0..7,16..23}; lanes 16-31: K={8..15,24..31}
__device__ __forceinline__ v16h load_b16(const _Float16* base, int ld, int lane) {
  int r  = lane & 15;
  int kh = (lane & 16) ? 8 : 0;
  HU u;
  u.h[0] = *(const v8h*)(base + r * ld + kh);
  u.h[1] = *(const v8h*)(base + r * ld + kh + 16);
  return u.v;
}

// ---------------- weight f32 -> f16 ----------------
__global__ __launch_bounds__(256) void cvt_f16_kernel(const float* __restrict__ s,
                                                      _Float16* __restrict__ d, int n) {
  int i = blockIdx.x * 256 + threadIdx.x;
  if (i < n) d[i] = (_Float16)s[i];
}

// ---------------- x[B][C][N] f32 -> xt[B][N][C] f16 (LDS tile transpose) ----------------
__global__ __launch_bounds__(256) void xpose_kernel(const float* __restrict__ x,
                                                    _Float16* __restrict__ xt) {
  __shared__ float tile[16][17];
  int b  = blockIdx.z;
  int c0 = blockIdx.x * 16;
  int n0 = blockIdx.y * 16;
  int tx = threadIdx.x, ty = threadIdx.y;
  tile[ty][tx] = x[((size_t)b * C + c0 + ty) * N + n0 + tx];
  __syncthreads();
  xt[((size_t)b * N + n0 + ty) * C + c0 + tx] = (_Float16)tile[tx][ty];
}

// ---------------- projections: F,G [B][N][64]; HT [B][512][N] ----------------
__global__ __launch_bounds__(128)
void proj_kernel(const _Float16* __restrict__ xt,
                 const _Float16* __restrict__ Wf, const _Float16* __restrict__ Wg,
                 const _Float16* __restrict__ Wh,
                 const float* __restrict__ bf, const float* __restrict__ bg,
                 const float* __restrict__ bh,
                 _Float16* __restrict__ F, _Float16* __restrict__ G,
                 _Float16* __restrict__ HT) {
  const int lane  = threadIdx.x & 31;
  const int w     = threadIdx.x >> 5;
  const int nbase = blockIdx.x * 16;
  const int chunk = blockIdx.y;   // 0: F|G (128 cols), 1..4: H cols
  const int b     = blockIdx.z;
  const _Float16* xb = xt + (size_t)b * N * C + (size_t)nbase * C;

  const _Float16* Wt; const float* bias; int col0; int target; // 0=F 1=G 2=HT
  if (chunk == 0) {
    if (w < 2) { Wt = Wf; bias = bf; col0 = w * 32;       target = 0; }
    else       { Wt = Wg; bias = bg; col0 = (w - 2) * 32; target = 1; }
  } else {
    Wt = Wh; bias = bh; col0 = (chunk - 1) * 128 + w * 32; target = 2;
  }

  v8f acc0 = {0.f,0.f,0.f,0.f,0.f,0.f,0.f,0.f};
  v8f acc1 = acc0;
  #pragma unroll 4
  for (int c = 0; c < C; c += 32) {
    v16h a  = load_a16(xb + c, C, lane);
    v16h b0 = load_b16(Wt + (size_t)col0 * C + c, C, lane);
    v16h b1 = load_b16(Wt + (size_t)(col0 + 16) * C + c, C, lane);
    acc0 = WMMA_F16(a, b0, acc0);
    acc1 = WMMA_F16(a, b1, acc1);
  }

  const int rh = (lane & 16) ? 8 : 0;  // row-half offset in D layout
  #pragma unroll
  for (int t = 0; t < 2; ++t) {
    v8f av = t ? acc1 : acc0;
    int col = col0 + t * 16 + (lane & 15);
    float bv = bias[col];
    if (target == 2) {
      v8h pk;
      #pragma unroll
      for (int j = 0; j < 8; ++j) pk[j] = (_Float16)(av[j] + bv);
      *(v8h*)(HT + (size_t)b * C * N + (size_t)col * N + nbase + rh) = pk;
    } else {
      _Float16* dst = (target == 0 ? F : G) + (size_t)b * N * CK
                      + (size_t)(nbase + rh) * CK + col;
      #pragma unroll
      for (int j = 0; j < 8; ++j) dst[(size_t)j * CK] = (_Float16)(av[j] + bv);
    }
  }
}

// ---------------- flash attention + fused epilogue (m-step = 64) ----------------
__global__ __launch_bounds__(128)
void attn_kernel(const _Float16* __restrict__ F, const _Float16* __restrict__ G,
                 const _Float16* __restrict__ HT, const float* __restrict__ x,
                 const float* __restrict__ gamma, float* __restrict__ out) {
  __shared__ _Float16 pbuf[4][16 * 64];   // per-wave private P buffer (no barriers)
  const int lane  = threadIdx.x & 31;
  const int w     = threadIdx.x >> 5;
  const int nbase = blockIdx.x * 16;      // 16 query rows per block
  const int b     = blockIdx.y;
  const int cbase = w * 128;              // 128 output channels per wave
  const int rh    = (lane & 16) ? 8 : 0;
  const _Float16* Fb = F + (size_t)b * N * CK;
  const _Float16* Gb = G + (size_t)b * N * CK;
  const _Float16* Hb = HT + (size_t)b * C * N;
  _Float16* pb = pbuf[w];

  const v8f vzero = {0.f,0.f,0.f,0.f,0.f,0.f,0.f,0.f};
  v16h fa0 = load_a16(Fb + (size_t)nbase * CK, CK, lane);        // K = 0..31
  v16h fa1 = load_a16(Fb + (size_t)nbase * CK + 32, CK, lane);   // K = 32..63

  v8f acc[8];
  #pragma unroll
  for (int t = 0; t < 8; ++t) acc[t] = vzero;
  float mr[8], lr[8];
  #pragma unroll
  for (int j = 0; j < 8; ++j) { mr[j] = -1e30f; lr[j] = 0.f; }

  #pragma unroll 1
  for (int m = 0; m < N; m += 64) {
    // ---- S = F @ G^T (16x64 tile, K=64): 4 sub-tiles, 8 WMMA ----
    v8f s[4];
    #pragma unroll
    for (int st = 0; st < 4; ++st) {
      const _Float16* gb = Gb + (size_t)(m + st * 16) * CK;
      v16h g0 = load_b16(gb, CK, lane);
      v16h g1 = load_b16(gb + 32, CK, lane);
      v8f t = vzero;
      t = WMMA_F16(fa0, g0, t);
      t = WMMA_F16(fa1, g1, t);
      s[st] = t;
    }

    // ---- online softmax (rows = D-layout VGPR index, cols = lanes within half) ----
    #pragma unroll
    for (int j = 0; j < 8; ++j) {
      float t = fmaxf(fmaxf(s[0][j], s[1][j]), fmaxf(s[2][j], s[3][j]));
      t = fmaxf(t, __shfl_xor(t, 1, 32));
      t = fmaxf(t, __shfl_xor(t, 2, 32));
      t = fmaxf(t, __shfl_xor(t, 4, 32));
      t = fmaxf(t, __shfl_xor(t, 8, 32));
      float mnew = fmaxf(mr[j], t);
      float sc = __expf(mr[j] - mnew);
      mr[j] = mnew;
      float p0 = __expf(s[0][j] - mnew);
      float p1 = __expf(s[1][j] - mnew);
      float p2 = __expf(s[2][j] - mnew);
      float p3 = __expf(s[3][j] - mnew);
      float rs = (p0 + p1) + (p2 + p3);
      rs += __shfl_xor(rs, 1, 32);
      rs += __shfl_xor(rs, 2, 32);
      rs += __shfl_xor(rs, 4, 32);
      rs += __shfl_xor(rs, 8, 32);
      lr[j] = lr[j] * sc + rs;
      #pragma unroll
      for (int t2 = 0; t2 < 8; ++t2) acc[t2][j] *= sc;
      int rowoff = (rh + j) * 64 + (lane & 15);      // D-layout -> row-major P in LDS
      pb[rowoff]      = (_Float16)p0;
      pb[rowoff + 16] = (_Float16)p1;
      pb[rowoff + 32] = (_Float16)p2;
      pb[rowoff + 48] = (_Float16)p3;
    }

    // ---- O += P @ H (P as A-operand from private LDS; 8 channel tiles, 16 WMMA) ----
    v16h pa0 = load_a16(pb, 64, lane);        // P cols m..m+31
    v16h pa1 = load_a16(pb + 32, 64, lane);   // P cols m+32..m+63
    #pragma unroll
    for (int t = 0; t < 8; ++t) {
      const _Float16* hbase = Hb + (size_t)(cbase + t * 16) * N + m;
      v16h hb0 = load_b16(hbase, N, lane);
      v16h hb1 = load_b16(hbase + 32, N, lane);
      acc[t] = WMMA_F16(pa0, hb0, acc[t]);
      acc[t] = WMMA_F16(pa1, hb1, acc[t]);
    }
  }

  // ---- out[b][c][n] = gamma * O/l + x  (contiguous 8-float runs per lane) ----
  float gm = gamma[0];
  float inv[8];
  #pragma unroll
  for (int j = 0; j < 8; ++j) inv[j] = gm / lr[j];
  #pragma unroll
  for (int t = 0; t < 8; ++t) {
    int c = cbase + t * 16 + (lane & 15);
    size_t off = ((size_t)b * C + c) * N + nbase + rh;
    const float* xp = x + off;
    float* op = out + off;
    #pragma unroll
    for (int j = 0; j < 8; ++j)
      op[j] = fmaf(acc[t][j], inv[j], xp[j]);
  }
}

extern "C" void kernel_launch(void* const* d_in, const int* in_sizes, int n_in,
                              void* d_out, int out_size, void* d_ws, size_t ws_size,
                              hipStream_t stream) {
  const float* x     = (const float*)d_in[0];
  const float* Wf_w  = (const float*)d_in[1];
  const float* Wf_b  = (const float*)d_in[2];
  const float* Wg_w  = (const float*)d_in[3];
  const float* Wg_b  = (const float*)d_in[4];
  const float* Wh_w  = (const float*)d_in[5];
  const float* Wh_b  = (const float*)d_in[6];
  const float* gamma = (const float*)d_in[7];
  float* out = (float*)d_out;

  // workspace layout (f16): xt | F | G | HT | Wf16 | Wg16 | Wh16  (~72.6 MB)
  _Float16* xt   = (_Float16*)d_ws;
  _Float16* F    = xt + (size_t)Bn * N * C;
  _Float16* G    = F + (size_t)Bn * N * CK;
  _Float16* HT   = G + (size_t)Bn * N * CK;
  _Float16* Wf16 = HT + (size_t)Bn * C * N;
  _Float16* Wg16 = Wf16 + CK * C;
  _Float16* Wh16 = Wg16 + CK * C;

  cvt_f16_kernel<<<(CK * C + 255) / 256, 256, 0, stream>>>(Wf_w, Wf16, CK * C);
  cvt_f16_kernel<<<(CK * C + 255) / 256, 256, 0, stream>>>(Wg_w, Wg16, CK * C);
  cvt_f16_kernel<<<(C * C + 255) / 256, 256, 0, stream>>>(Wh_w, Wh16, C * C);
  xpose_kernel<<<dim3(C / 16, N / 16, Bn), dim3(16, 16), 0, stream>>>(x, xt);
  proj_kernel<<<dim3(N / 16, 5, Bn), 128, 0, stream>>>(xt, Wf16, Wg16, Wh16,
                                                       Wf_b, Wg_b, Wh_b, F, G, HT);
  attn_kernel<<<dim3(N / 16, Bn), 128, 0, stream>>>(F, G, HT, x, gamma, out);
}
